// SequenceDecoder_15393162788885
// MI455X (gfx1250) — matrix operand order repaired and checked by reading, not compiled, not run on hardware
//
#include <hip/hip_runtime.h>
#include <stdint.h>

#define B_   128
#define EMB_ 512
#define H_   1024
#define V_   8192
#define T_   100

typedef __attribute__((ext_vector_type(16))) __bf16 v16bf;
typedef __attribute__((ext_vector_type(8)))  float  v8f;
typedef __attribute__((ext_vector_type(4)))  unsigned int v4u;

union ABFrag { v16bf v; v4u q[2]; };

__device__ __forceinline__ unsigned short f2bf(float f) {
  union { float f; unsigned int u; } c; c.f = f;
  unsigned int u = c.u;
  return (unsigned short)((u + 0x7FFFu + ((u >> 16) & 1u)) >> 16);
}

// A fragment (16x32 bf16, row-major source): lane supplies row (lane&15);
// hi half-wave (lane>=16) holds K pattern offset by 8 (ISA 7.12.2 layout).
__device__ __forceinline__ v16bf load_a16x32(const unsigned short* __restrict__ row,
                                             int k0, int hi) {
  ABFrag u;
  u.q[0] = *(const v4u*)(row + k0 + hi * 8);
  u.q[1] = *(const v4u*)(row + k0 + 16 + hi * 8);
  return u.v;
}

// B fragment (32x16 bf16 = W^T tile): lane supplies output column (lane&15),
// i.e. one row of W; hi half-wave holds K pattern offset by 8 (same as A).
__device__ __forceinline__ v16bf load_b32x16(const unsigned short* __restrict__ wrow,
                                             int k0, int hi) {
  ABFrag u;
  const unsigned short* p = wrow + k0 + hi * 8;
  u.q[0] = *(const v4u*)(p);
  u.q[1] = *(const v4u*)(p + 16);
  return u.v;
}

__device__ __forceinline__ v8f wmma_bf16(v16bf a, v16bf b, v8f c) {
  return __builtin_amdgcn_wmma_f32_16x16x32_bf16(false, a, false, b, (short)0, c,
                                                 false, false);
}

__device__ __forceinline__ float sigmoidf_fast(float x) {
  return 1.0f / (1.0f + __expf(-x));
}
__device__ __forceinline__ float tanhf_fast(float x) {
  float e2 = __expf(2.0f * x);
  return (e2 - 1.0f) / (e2 + 1.0f);
}

// ---------------- prep kernels ----------------

__global__ void k_f32_to_bf16(const float* __restrict__ src,
                              unsigned short* __restrict__ dst, int n) {
  int i = blockIdx.x * blockDim.x + threadIdx.x;
  if (i < n) dst[i] = f2bf(src[i]);
}

__global__ void k_init_h(const float* __restrict__ x, float* __restrict__ hf,
                         unsigned short* __restrict__ hb, int n) {
  int i = blockIdx.x * blockDim.x + threadIdx.x;
  if (i < n) { float v = x[i]; hf[i] = v; hb[i] = f2bf(v); }
}

// t=0 plane: log_softmax(one-hot at 0): max=1, lse = log(1 + 8191*e^-1)
__global__ void k_fill_t0(float* __restrict__ out, int* __restrict__ idx) {
  int i = blockIdx.x * blockDim.x + threadIdx.x;
  float lse = __logf(1.0f + 8191.0f * __expf(-1.0f));
  if (i < B_ * V_) {
    int b = i >> 13;          // /V_
    int v = i & (V_ - 1);
    out[((size_t)b * T_) * V_ + v] = (v == 0) ? -lse : (-1.0f - lse);
  }
  if (i < B_) idx[i] = 0;
}

// ---------------- per-step kernels ----------------

__global__ void k_gather(const int* __restrict__ idx, const float* __restrict__ emb,
                         unsigned short* __restrict__ e_bf) {
  int b = blockIdx.x;
  const float* src = emb + (size_t)idx[b] * EMB_;
  for (int i = threadIdx.x; i < EMB_; i += blockDim.x)
    e_bf[b * EMB_ + i] = f2bf(src[i]);
}

// Fused GRU cell: gi = e@Wih^T, gh = h@Whh^T, gates, h_new.
// 512 waves: 8 row tiles (B/16) x 64 col tiles (H/16), 6 accumulators each.
__global__ __launch_bounds__(256) void k_gru(
    const unsigned short* __restrict__ e_bf,   // [B, EMB]
    const unsigned short* __restrict__ h_bf,   // [B, H]
    const float* __restrict__ h_f32,           // [B, H]
    const unsigned short* __restrict__ wih_bf, // [3H, EMB]
    const unsigned short* __restrict__ whh_bf, // [3H, H]
    const float* __restrict__ b_ih, const float* __restrict__ b_hh,
    float* __restrict__ hf_out, unsigned short* __restrict__ hb_out) {
  const int lane = threadIdx.x & 31;
  const int wave = blockIdx.x * (blockDim.x >> 5) + (threadIdx.x >> 5);
  const int rowTile = wave >> 6;   // 0..7
  const int colTile = wave & 63;   // 0..63
  const int hi = lane >> 4;
  const int lc = lane & 15;
  const int m0 = rowTile * 16;
  const int ncol = colTile * 16 + lc;

  v8f acc_ir = {}, acc_iz = {}, acc_in = {};
  v8f acc_hr = {}, acc_hz = {}, acc_hn = {};

  // input-gate GEMM over EMB
  {
    const unsigned short* arow = e_bf + (m0 + lc) * EMB_;
    const unsigned short* br = wih_bf + (size_t)(0 * H_ + ncol) * EMB_;
    const unsigned short* bz = wih_bf + (size_t)(1 * H_ + ncol) * EMB_;
    const unsigned short* bn = wih_bf + (size_t)(2 * H_ + ncol) * EMB_;
#pragma unroll 4
    for (int k = 0; k < EMB_; k += 32) {
      v16bf a = load_a16x32(arow, k, hi);
      acc_ir = wmma_bf16(a, load_b32x16(br, k, hi), acc_ir);
      acc_iz = wmma_bf16(a, load_b32x16(bz, k, hi), acc_iz);
      acc_in = wmma_bf16(a, load_b32x16(bn, k, hi), acc_in);
    }
  }
  // hidden-gate GEMM over H
  {
    const unsigned short* arow = h_bf + (m0 + lc) * H_;
    const unsigned short* br = whh_bf + (size_t)(0 * H_ + ncol) * H_;
    const unsigned short* bz = whh_bf + (size_t)(1 * H_ + ncol) * H_;
    const unsigned short* bn = whh_bf + (size_t)(2 * H_ + ncol) * H_;
#pragma unroll 4
    for (int k = 0; k < H_; k += 32) {
      v16bf a = load_a16x32(arow, k, hi);
      acc_hr = wmma_bf16(a, load_b32x16(br, k, hi), acc_hr);
      acc_hz = wmma_bf16(a, load_b32x16(bz, k, hi), acc_hz);
      acc_hn = wmma_bf16(a, load_b32x16(bn, k, hi), acc_hn);
    }
  }

  const float bihr = b_ih[ncol], bihz = b_ih[H_ + ncol], bihn = b_ih[2 * H_ + ncol];
  const float bhhr = b_hh[ncol], bhhz = b_hh[H_ + ncol], bhhn = b_hh[2 * H_ + ncol];
#pragma unroll
  for (int i = 0; i < 8; ++i) {
    int m = m0 + i + 8 * hi;   // C/D layout: row = vgpr + 8*(lane>=16)
    float r = sigmoidf_fast((acc_ir[i] + bihr) + (acc_hr[i] + bhhr));
    float z = sigmoidf_fast((acc_iz[i] + bihz) + (acc_hz[i] + bhhz));
    float n = tanhf_fast((acc_in[i] + bihn) + r * (acc_hn[i] + bhhn));
    float hp = h_f32[m * H_ + ncol];
    float hn = (1.0f - z) * n + z * hp;
    hf_out[m * H_ + ncol] = hn;
    hb_out[m * H_ + ncol] = f2bf(hn);
  }
}

// logits = h @ cls_w^T + cls_b. Each wave: one 16-row tile x FOUR 16-col tiles
// (shared A fragment across 4 B tiles -> 2.5 loads per WMMA instead of 4).
// 1024 waves: 8 row tiles x 128 col groups (64 cols each). 128 blocks x 8 waves.
__global__ __launch_bounds__(256) void k_logits(
    const unsigned short* __restrict__ h_bf,    // [B, H]
    const unsigned short* __restrict__ clsw_bf, // [V, H]
    const float* __restrict__ cls_b, float* __restrict__ logits) {
  const int lane = threadIdx.x & 31;
  const int wave = blockIdx.x * (blockDim.x >> 5) + (threadIdx.x >> 5);
  const int rowTile = wave >> 7;    // 0..7
  const int colGrp  = wave & 127;   // 0..127 (64 columns each)
  const int hi = lane >> 4;
  const int lc = lane & 15;
  const int m0 = rowTile * 16;
  const int n0 = colGrp * 64;

  v8f acc0 = {}, acc1 = {}, acc2 = {}, acc3 = {};
  const unsigned short* arow = h_bf + (m0 + lc) * H_;
  const unsigned short* b0 = clsw_bf + (size_t)(n0 + 0  + lc) * H_;
  const unsigned short* b1 = clsw_bf + (size_t)(n0 + 16 + lc) * H_;
  const unsigned short* b2 = clsw_bf + (size_t)(n0 + 32 + lc) * H_;
  const unsigned short* b3 = clsw_bf + (size_t)(n0 + 48 + lc) * H_;
#pragma unroll 4
  for (int k = 0; k < H_; k += 32) {
    // prefetch next K-chunk of the streaming weight rows (global_prefetch path)
    if (k + 32 < H_) {
      __builtin_prefetch(b0 + k + 32, 0, 1);
      __builtin_prefetch(b1 + k + 32, 0, 1);
      __builtin_prefetch(b2 + k + 32, 0, 1);
      __builtin_prefetch(b3 + k + 32, 0, 1);
    }
    v16bf a = load_a16x32(arow, k, hi);
    acc0 = wmma_bf16(a, load_b32x16(b0, k, hi), acc0);
    acc1 = wmma_bf16(a, load_b32x16(b1, k, hi), acc1);
    acc2 = wmma_bf16(a, load_b32x16(b2, k, hi), acc2);
    acc3 = wmma_bf16(a, load_b32x16(b3, k, hi), acc3);
  }
  const float bias0 = cls_b[n0 + 0  + lc];
  const float bias1 = cls_b[n0 + 16 + lc];
  const float bias2 = cls_b[n0 + 32 + lc];
  const float bias3 = cls_b[n0 + 48 + lc];
#pragma unroll
  for (int i = 0; i < 8; ++i) {
    const size_t m = (size_t)(m0 + i + 8 * hi);
    logits[m * V_ + n0 + 0  + lc] = acc0[i] + bias0;
    logits[m * V_ + n0 + 16 + lc] = acc1[i] + bias1;
    logits[m * V_ + n0 + 32 + lc] = acc2[i] + bias2;
    logits[m * V_ + n0 + 48 + lc] = acc3[i] + bias3;
  }
}

// Fused argmax (feeds next step) + log_softmax row write into d_out[:, t, :].
__global__ __launch_bounds__(256) void k_softmax(
    const float* __restrict__ logits, float* __restrict__ out,
    int* __restrict__ idx, int t) {
  __shared__ float smax[256];
  __shared__ int   sarg[256];
  __shared__ float ssum[256];
  const int b = blockIdx.x, tid = threadIdx.x;
  const float* row = logits + (size_t)b * V_;

  float best = -3.402823466e38f; int bi = V_;
  for (int v = tid; v < V_; v += 256) {
    float x = row[v];
    if (x > best) { best = x; bi = v; }
  }
  smax[tid] = best; sarg[tid] = bi;
  __syncthreads();
  for (int s = 128; s > 0; s >>= 1) {
    if (tid < s) {
      float o = smax[tid + s]; int oi = sarg[tid + s];
      if (o > smax[tid] || (o == smax[tid] && oi < sarg[tid])) {
        smax[tid] = o; sarg[tid] = oi;
      }
    }
    __syncthreads();
  }
  const float rmax = smax[0];
  if (tid == 0) idx[b] = sarg[0];

  float acc = 0.0f;
  for (int v = tid; v < V_; v += 256) acc += __expf(row[v] - rmax);
  ssum[tid] = acc;
  __syncthreads();
  for (int s = 128; s > 0; s >>= 1) {
    if (tid < s) ssum[tid] += ssum[tid + s];
    __syncthreads();
  }
  const float sub = rmax + __logf(ssum[0]);
  float* orow = out + ((size_t)b * T_ + t) * V_;
  for (int v = tid; v < V_; v += 256) orow[v] = row[v] - sub;
}

// ---------------- host ----------------

extern "C" void kernel_launch(void* const* d_in, const int* in_sizes, int n_in,
                              void* d_out, int out_size, void* d_ws, size_t ws_size,
                              hipStream_t stream) {
  (void)in_sizes; (void)n_in; (void)out_size; (void)ws_size;
  const float* x     = (const float*)d_in[0];
  const float* emb   = (const float*)d_in[1];
  const float* w_ih  = (const float*)d_in[2];
  const float* w_hh  = (const float*)d_in[3];
  const float* b_ih  = (const float*)d_in[4];
  const float* b_hh  = (const float*)d_in[5];
  const float* cls_w = (const float*)d_in[6];
  const float* cls_b = (const float*)d_in[7];
  float* out = (float*)d_out;

  char* ws = (char*)d_ws;
  size_t off = 0;
  auto alloc = [&](size_t bytes) -> void* {
    void* p = ws + off;
    off += (bytes + 255) & ~(size_t)255;
    return p;
  };
  unsigned short* wih_bf  = (unsigned short*)alloc((size_t)3 * H_ * EMB_ * 2);
  unsigned short* whh_bf  = (unsigned short*)alloc((size_t)3 * H_ * H_ * 2);
  unsigned short* clsw_bf = (unsigned short*)alloc((size_t)V_ * H_ * 2);
  float* hf0 = (float*)alloc((size_t)B_ * H_ * 4);
  float* hf1 = (float*)alloc((size_t)B_ * H_ * 4);
  unsigned short* hb0 = (unsigned short*)alloc((size_t)B_ * H_ * 2);
  unsigned short* hb1 = (unsigned short*)alloc((size_t)B_ * H_ * 2);
  unsigned short* e_bf = (unsigned short*)alloc((size_t)B_ * EMB_ * 2);
  float* logits = (float*)alloc((size_t)B_ * V_ * 4);
  int* idx = (int*)alloc(B_ * 4);
  float* hf[2] = {hf0, hf1};
  unsigned short* hb[2] = {hb0, hb1};

  int n;
  n = 3 * H_ * EMB_; k_f32_to_bf16<<<(n + 255) / 256, 256, 0, stream>>>(w_ih, wih_bf, n);
  n = 3 * H_ * H_;   k_f32_to_bf16<<<(n + 255) / 256, 256, 0, stream>>>(w_hh, whh_bf, n);
  n = V_ * H_;       k_f32_to_bf16<<<(n + 255) / 256, 256, 0, stream>>>(cls_w, clsw_bf, n);
  n = B_ * H_;       k_init_h<<<(n + 255) / 256, 256, 0, stream>>>(x, hf[0], hb[0], n);
  k_fill_t0<<<(B_ * V_ + 255) / 256, 256, 0, stream>>>(out, idx);

  for (int t = 1; t < T_; ++t) {
    const int src = (t - 1) & 1, dst = t & 1;
    k_gather<<<B_, 128, 0, stream>>>(idx, emb, e_bf);
    k_gru<<<64, 256, 0, stream>>>(e_bf, hb[src], hf[src], wih_bf, whh_bf,
                                  b_ih, b_hh, hf[dst], hb[dst]);
    k_logits<<<128, 256, 0, stream>>>(hb[dst], clsw_bf, cls_b, logits);
    k_softmax<<<B_, 256, 0, stream>>>(logits, out, idx, t);
  }
}
